// MambaBlock_27255862460953
// MI455X (gfx1250) — matrix operand order repaired and checked by reading, not compiled, not run on hardware
//
#include <hip/hip_runtime.h>
#include <hip/hip_bf16.h>

// ---------------- types for WMMA fragments ----------------
typedef __attribute__((ext_vector_type(16))) __bf16       v16bf;
typedef __attribute__((ext_vector_type(8)))  float        v8f;
typedef __attribute__((ext_vector_type(4)))  unsigned int u32x4;  // builtin vec: AS-safe

union FragU {
    u32x4 q[2];
    v16bf v;
};

// explicit global-address-space pointers: force global_load/global_store
typedef const __attribute__((address_space(1))) u32x4*          gld_u4;
typedef const __attribute__((address_space(1))) float*          gld_f;
typedef const __attribute__((address_space(1))) unsigned short* gld_us;
typedef __attribute__((address_space(1))) float*                gst_f;
typedef __attribute__((address_space(1))) unsigned short*       gst_us;

#define D_MODEL   1024
#define D_INNER   2048
#define D_STATE   16
#define DT_RANK   64
#define SEQ_L     2048
#define NBATCH    2
#define BL_TOTAL  (NBATCH * SEQ_L)          // 4096 rows

__device__ __forceinline__ unsigned short f32_to_bf16(float f) {
    unsigned u = __float_as_uint(f);
    unsigned rounding = 0x7FFFu + ((u >> 16) & 1u);   // round-to-nearest-even
    u += rounding;
    return (unsigned short)(u >> 16);
}

// ---------------- fp32 -> bf16 convert ----------------
__global__ void cvt_bf16_kernel(const float* __restrict__ src,
                                unsigned short* __restrict__ dst, int n) {
    int i = blockIdx.x * blockDim.x + threadIdx.x;
    if (i < n) ((gst_us)dst)[i] = f32_to_bf16(((gld_f)src)[i]);
}

// ---------------- DynamicTanh norm -> bf16 ----------------
__global__ void norm_kernel(const float* __restrict__ x,
                            const float* __restrict__ alpha,
                            const float* __restrict__ w,
                            const float* __restrict__ b,
                            unsigned short* __restrict__ out, int n) {
    int i = blockIdx.x * blockDim.x + threadIdx.x;
    if (i >= n) return;
    float a = fminf(fmaxf(((gld_f)alpha)[0], 0.1f), 2.0f);
    int d = i & (D_MODEL - 1);
    float v = ((gld_f)w)[d] * tanhf(a * ((gld_f)x)[i]) + ((gld_f)b)[d];
    ((gst_us)out)[i] = f32_to_bf16(v);
}

// ---------------- depthwise causal conv1d + SiLU ----------------
__global__ void conv_silu_kernel(const float* __restrict__ xz,
                                 const float* __restrict__ cw,
                                 const float* __restrict__ cb,
                                 float* __restrict__ u_f,
                                 unsigned short* __restrict__ u_bf, int total) {
    int i = blockIdx.x * blockDim.x + threadIdx.x;
    if (i >= total) return;
    int e  = i & (D_INNER - 1);
    int bl = i >> 11;                       // D_INNER == 2048
    int l  = bl & (SEQ_L - 1);
    float s = ((gld_f)cb)[e];
#pragma unroll
    for (int j = 0; j < 4; ++j) {
        int lj = l - 3 + j;
        if (lj >= 0)
            s += ((gld_f)cw)[e * 4 + j] *
                 ((gld_f)xz)[(size_t)(bl - 3 + j) * (2 * D_INNER) + e];
    }
    float r = s / (1.0f + __expf(-s));      // SiLU
    ((gst_f)u_f)[i]  = r;
    ((gst_us)u_bf)[i] = f32_to_bf16(r);
}

// ---------------- generic bf16 WMMA GEMM: C = A(MxK) * W(NxK)^T ----------------
// MT x NT 16x16 tiles per wave (32x64 output for MT=2,NT=4).
// Software-pipelined K loop (ping-pong fragment buffers, K % 64 == 0).
// MODE 0: plain fp32 store
// MODE 1: fp32 store + bf16 copy of columns n<64 into aux (dt_lr)
// MODE 2: fused softplus(val + bias[n])
template <int MT, int NT, int MODE>
__global__ __launch_bounds__(256) void gemm_bf16_kernel(
    const unsigned short* __restrict__ A, const unsigned short* __restrict__ Bw,
    float* __restrict__ C, int M, int N, int K, int lda, int ldb, int ldc,
    const float* __restrict__ bias, unsigned short* __restrict__ aux, int auxLd) {
    const int lane = threadIdx.x & 31;
    const int gid  = blockIdx.x * 8 + (threadIdx.x >> 5);
    const int mtiles = M / (MT * 16);
    const int mt = gid % mtiles;
    const int ng = gid / mtiles;
    const int m0 = mt * (MT * 16);
    const int n0 = ng * (NT * 16);

    v8f acc[MT][NT];
#pragma unroll
    for (int mi = 0; mi < MT; ++mi)
#pragma unroll
        for (int t = 0; t < NT; ++t)
#pragma unroll
            for (int v = 0; v < 8; ++v) acc[mi][t][v] = 0.0f;

    // A fragment (16x32 bf16): lanes 0-15 read K+{0..7},K+{16..23}; lanes 16-31 +8
    gld_us ap[MT];
#pragma unroll
    for (int mi = 0; mi < MT; ++mi)
        ap[mi] = (gld_us)A + (size_t)(m0 + mi * 16 + (lane & 15)) * lda +
                 ((lane < 16) ? 0 : 8);
    // B fragment (32x16): lane n = lane&15, lanes 0-15 read K+{0..7},K+{16..23}; lanes 16-31 +8
    gld_us bp[NT];
#pragma unroll
    for (int t = 0; t < NT; ++t)
        bp[t] = (gld_us)Bw + (size_t)(n0 + t * 16 + (lane & 15)) * ldb +
                ((lane < 16) ? 0 : 8);

#define LOAD_STEP(abuf, bbuf)                                                  \
    {                                                                          \
        _Pragma("unroll") for (int mi = 0; mi < MT; ++mi) {                    \
            abuf[mi].q[0] = *(gld_u4)(ap[mi]);                                 \
            abuf[mi].q[1] = *(gld_u4)(ap[mi] + 16);                            \
            ap[mi] += 32;                                                      \
        }                                                                      \
        _Pragma("unroll") for (int t = 0; t < NT; ++t) {                       \
            bbuf[t].q[0] = *(gld_u4)(bp[t]);                                   \
            bbuf[t].q[1] = *(gld_u4)(bp[t] + 16);                              \
            bp[t] += 32;                                                       \
        }                                                                      \
    }

#define MMA_STEP(abuf, bbuf)                                                   \
    {                                                                          \
        _Pragma("unroll") for (int mi = 0; mi < MT; ++mi)                      \
            _Pragma("unroll") for (int t = 0; t < NT; ++t)                     \
                acc[mi][t] = __builtin_amdgcn_wmma_f32_16x16x32_bf16(          \
                    false, abuf[mi].v, false, bbuf[t].v, (short)0,             \
                    acc[mi][t], false, false);                                 \
    }

    FragU a0[MT], b0[NT], a1[MT], b1[NT];
    LOAD_STEP(a0, b0);                       // k-step 0
    for (int kb = 0; kb < K; kb += 64) {
        LOAD_STEP(a1, b1);                   // k-step kb+32 in flight over MMAs
        MMA_STEP(a0, b0);
        if (kb + 64 < K) LOAD_STEP(a0, b0);  // k-step kb+64 in flight over MMAs
        MMA_STEP(a1, b1);
    }
#undef LOAD_STEP
#undef MMA_STEP

    // C/D layout: VGPR v -> M = m0 + 16*mi + v + (lane<16?0:8), N = n0+16t+lane%16
#pragma unroll
    for (int mi = 0; mi < MT; ++mi) {
        const int rbase = m0 + mi * 16 + ((lane < 16) ? 0 : 8);
#pragma unroll
        for (int t = 0; t < NT; ++t) {
            const int n = n0 + t * 16 + (lane & 15);
            const bool dt_tile = (MODE == 1) && ((n0 + t * 16) < DT_RANK);
#pragma unroll
            for (int v = 0; v < 8; ++v) {
                int r = rbase + v;
                float val = acc[mi][t][v];
                if (MODE == 2) {
                    float x = val + ((gld_f)bias)[n];
                    val = (x > 20.0f) ? x : log1pf(__expf(x));   // softplus
                }
                ((gst_f)C)[(size_t)r * ldc + n] = val;
                if (dt_tile)
                    ((gst_us)aux)[(size_t)r * auxLd + n] = f32_to_bf16(val);
            }
        }
    }
}

// ---------------- selective scan (one lane per (b, e, s)) ----------------
__global__ __launch_bounds__(256) void scan_kernel(
    const float* __restrict__ dt, const float* __restrict__ u_f,
    const float* __restrict__ dbl, const float* __restrict__ xz,
    const float* __restrict__ A_log, const float* __restrict__ Dp,
    unsigned short* __restrict__ yg) {
    int t = blockIdx.x * blockDim.x + threadIdx.x;   // 0..65535
    int s = t & (D_STATE - 1);
    int e = (t >> 4) & (D_INNER - 1);
    int b = t >> 15;

    float Aes = -__expf(((gld_f)A_log)[e * D_STATE + s]);
    float Dpe = ((gld_f)Dp)[e];
    float h = 0.0f;
    const int baseBL = b * SEQ_L;
    gld_f dtp = (gld_f)dt + (size_t)baseBL * D_INNER + e;
    gld_f up  = (gld_f)u_f + (size_t)baseBL * D_INNER + e;
    gld_f Bp  = (gld_f)dbl + (size_t)baseBL * 96 + DT_RANK + s;
    gld_f Cp  = Bp + D_STATE;
    gld_f zp  = (gld_f)xz + (size_t)baseBL * (2 * D_INNER) + D_INNER + e;
    gst_us yp = (gst_us)yg + (size_t)baseBL * D_INNER + e;

    for (int l = 0; l < SEQ_L; ++l) {
        float dtv = dtp[(size_t)l * D_INNER];
        float uv  = up[(size_t)l * D_INNER];
        float Bv  = Bp[(size_t)l * 96];
        float Cv  = Cp[(size_t)l * 96];
        float dA  = __expf(dtv * Aes);
        h = dA * h + dtv * Bv * uv;
        float y = h * Cv;
        // reduce over the 16 states (stays inside each 16-lane half of wave32)
        y += __shfl_xor(y, 1);
        y += __shfl_xor(y, 2);
        y += __shfl_xor(y, 4);
        y += __shfl_xor(y, 8);
        if (s == 0) {
            float yf = y + uv * Dpe;
            float z  = zp[(size_t)l * (2 * D_INNER)];
            float g  = z / (1.0f + __expf(-z));
            yp[(size_t)l * D_INNER] = f32_to_bf16(yf * g);
        }
    }
}

// ---------------- host launcher ----------------
extern "C" void kernel_launch(void* const* d_in, const int* in_sizes, int n_in,
                              void* d_out, int out_size, void* d_ws, size_t ws_size,
                              hipStream_t stream) {
    (void)in_sizes; (void)n_in; (void)out_size; (void)ws_size;
    const float* x_flat    = (const float*)d_in[0];
    const float* alpha     = (const float*)d_in[1];
    const float* norm_w    = (const float*)d_in[2];
    const float* norm_b    = (const float*)d_in[3];
    const float* in_proj_w = (const float*)d_in[4];
    const float* conv_w    = (const float*)d_in[5];
    const float* conv_b    = (const float*)d_in[6];
    const float* x_proj_w  = (const float*)d_in[7];
    const float* dt_proj_w = (const float*)d_in[8];
    const float* dt_proj_b = (const float*)d_in[9];
    const float* A_log     = (const float*)d_in[10];
    const float* D_param   = (const float*)d_in[11];
    const float* out_proj_w= (const float*)d_in[12];
    float* out = (float*)d_out;

    char* ws = (char*)d_ws;
    size_t off = 0;
    auto alloc = [&](size_t bytes) -> void* {
        void* p = ws + off;
        off += (bytes + 255) & ~(size_t)255;
        return p;
    };
    unsigned short* w_in_bf  = (unsigned short*)alloc((size_t)2 * D_INNER * D_MODEL * 2);
    unsigned short* w_xp_bf  = (unsigned short*)alloc((size_t)96 * D_INNER * 2);
    unsigned short* w_dt_bf  = (unsigned short*)alloc((size_t)D_INNER * DT_RANK * 2);
    unsigned short* w_out_bf = (unsigned short*)alloc((size_t)D_MODEL * D_INNER * 2);
    unsigned short* xn_bf    = (unsigned short*)alloc((size_t)BL_TOTAL * D_MODEL * 2);
    float*          xz_f     = (float*)alloc((size_t)BL_TOTAL * 2 * D_INNER * 4);
    float*          u_f      = (float*)alloc((size_t)BL_TOTAL * D_INNER * 4);
    unsigned short* u_bf     = (unsigned short*)alloc((size_t)BL_TOTAL * D_INNER * 2);
    float*          dbl_f    = (float*)alloc((size_t)BL_TOTAL * 96 * 4);
    unsigned short* dtlr_bf  = (unsigned short*)alloc((size_t)BL_TOTAL * DT_RANK * 2);
    float*          dt_f     = (float*)alloc((size_t)BL_TOTAL * D_INNER * 4);
    unsigned short* yg_bf    = (unsigned short*)alloc((size_t)BL_TOTAL * D_INNER * 2);

    // 1) weights -> bf16
    {
        int n;
        n = 2 * D_INNER * D_MODEL;
        cvt_bf16_kernel<<<(n + 255) / 256, 256, 0, stream>>>(in_proj_w, w_in_bf, n);
        n = 96 * D_INNER;
        cvt_bf16_kernel<<<(n + 255) / 256, 256, 0, stream>>>(x_proj_w, w_xp_bf, n);
        n = D_INNER * DT_RANK;
        cvt_bf16_kernel<<<(n + 255) / 256, 256, 0, stream>>>(dt_proj_w, w_dt_bf, n);
        n = D_MODEL * D_INNER;
        cvt_bf16_kernel<<<(n + 255) / 256, 256, 0, stream>>>(out_proj_w, w_out_bf, n);
    }

    // 2) DynTanh norm -> bf16
    {
        int n = BL_TOTAL * D_MODEL;
        norm_kernel<<<(n + 255) / 256, 256, 0, stream>>>(x_flat, alpha, norm_w, norm_b, xn_bf, n);
    }

    // 3) in_proj GEMM: xz(4096x4096) = xn(4096x1024) * in_proj_w^T
    {
        int M = BL_TOTAL, N = 2 * D_INNER, K = D_MODEL;
        int waves = (M / 32) * (N / 64);           // 8192
        gemm_bf16_kernel<2, 4, 0><<<waves / 8, 256, 0, stream>>>(
            xn_bf, w_in_bf, xz_f, M, N, K, K, K, N, nullptr, nullptr, 0);
    }

    // 4) depthwise conv + SiLU
    {
        int n = BL_TOTAL * D_INNER;
        conv_silu_kernel<<<(n + 255) / 256, 256, 0, stream>>>(xz_f, conv_w, conv_b, u_f, u_bf, n);
    }

    // 5) x_proj GEMM: dbl(4096x96) = u(4096x2048) * x_proj_w^T  (NT=2: 3 groups of 32)
    {
        int M = BL_TOTAL, N = 96, K = D_INNER;
        int waves = (M / 32) * (N / 32);           // 384
        gemm_bf16_kernel<2, 2, 1><<<waves / 8, 256, 0, stream>>>(
            u_bf, w_xp_bf, dbl_f, M, N, K, K, K, N, nullptr, dtlr_bf, DT_RANK);
    }

    // 6) dt GEMM: dt(4096x2048) = softplus(dt_lr(4096x64) * dt_proj_w^T + b)
    {
        int M = BL_TOTAL, N = D_INNER, K = DT_RANK;
        int waves = (M / 32) * (N / 64);           // 4096
        gemm_bf16_kernel<2, 4, 2><<<waves / 8, 256, 0, stream>>>(
            dtlr_bf, w_dt_bf, dt_f, M, N, K, K, K, N, dt_proj_b, nullptr, 0);
    }

    // 7) selective scan + gating epilogue -> bf16
    {
        int n = NBATCH * D_INNER * D_STATE;        // 65536 lanes
        scan_kernel<<<n / 256, 256, 0, stream>>>(dt_f, u_f, dbl_f, xz_f, A_log, D_param, yg_bf);
    }

    // 8) out_proj GEMM: out(4096x1024) = yg(4096x2048) * out_proj_w^T
    {
        int M = BL_TOTAL, N = D_MODEL, K = D_INNER;
        int waves = (M / 32) * (N / 64);           // 2048
        gemm_bf16_kernel<2, 4, 0><<<waves / 8, 256, 0, stream>>>(
            yg_bf, w_out_bf, out, M, N, K, K, K, N, nullptr, nullptr, 0);
    }
}
